// FlashAttention_23854248362849
// MI455X (gfx1250) — matrix operand order repaired and checked by reading, not compiled, not run on hardware
//
#include <hip/hip_runtime.h>
#include <math.h>

typedef __attribute__((ext_vector_type(16))) _Float16     v16h;
typedef __attribute__((ext_vector_type(8)))  _Float16     v8h;
typedef __attribute__((ext_vector_type(8)))  float        v8f;
typedef __attribute__((ext_vector_type(4)))  unsigned int u32x4;
typedef __attribute__((ext_vector_type(8)))  int          i32x8;
typedef __attribute__((ext_vector_type(4)))  int          i32x4;

#define SEQ   2048
#define HEADS 16
#define HDIM  128
#define HID   2048
#define QT    16            // query rows per wave
#define KT    64            // keys per kv iteration
#define NT    (KT/16)       // score n-tiles per iteration (4)
#define WAVES 8
#define QBLK  (QT*WAVES)    // 128 query rows per block

__global__ __launch_bounds__(256)
void fa_fwd_f16wmma(const float* __restrict__ Q,
                    const float* __restrict__ K,
                    const float* __restrict__ V,
                    const float* __restrict__ mask,
                    float* __restrict__ out) {
    __shared__ __align__(32) _Float16 Ksh[KT * HDIM];        // 64 keys x 128 d (row-major)   16KB
    __shared__ __align__(32) _Float16 Vsh[HDIM * KT];        // 128 d x 64 keys (transposed)  16KB
    __shared__ __align__(32) float    Msh[QBLK * KT];        // mask tile (TDM target)        32KB
    __shared__ __align__(32) _Float16 Psh[WAVES][QT * KT];   // per-wave P scratch            16KB

    const int tid   = threadIdx.x;
    const int wave  = tid >> 5;
    const int lane  = tid & 31;
    const int n16   = lane & 15;   // N column / A-row index
    const int hhalf = lane >> 4;   // lane half selects K/M sub-range

    const int nQblk = SEQ / QBLK;                 // 16
    const int qblk  = blockIdx.x % nQblk;
    const int head  = (blockIdx.x / nQblk) % HEADS;
    const int batch =  blockIdx.x / (nQblk * HEADS);

    const int   q0    = qblk * QBLK + wave * QT;
    const float scale = 0.08838834764831845f;     // 1/sqrt(128)

    // ---- Q tile -> f16 A-fragments in registers (4 chunks of K=32) ----
    v16h qf[4];
    {
        const float* qrow = Q + ((size_t)(batch * SEQ + q0 + n16)) * HID + head * HDIM;
        #pragma unroll
        for (int c = 0; c < 4; ++c) {
            const int d0 = c * 32 + hhalf * 16;   // A layout: low 8 halves
            const int d1 = d0 + 8;                // high 8 halves
            v16h a = {};
            #pragma unroll
            for (int i = 0; i < 8; ++i) a[i]     = (_Float16)(qrow[d0 + i] * scale);
            #pragma unroll
            for (int i = 0; i < 8; ++i) a[8 + i] = (_Float16)(qrow[d1 + i] * scale);
            qf[c] = a;
        }
    }

    // ones B-matrix for WMMA-based row sums
    v16h vones;
    #pragma unroll
    for (int i = 0; i < 16; ++i) vones[i] = (_Float16)1.0f;

    // ---- flash accumulators ----
    v8f o[8];
    #pragma unroll
    for (int t = 0; t < 8; ++t) o[t] = {};
    float mrow[8], lrow[8];
    #pragma unroll
    for (int r = 0; r < 8; ++r) { mrow[r] = -1e30f; lrow[r] = 0.0f; }

    const size_t   kvBase  = ((size_t)batch * SEQ) * HID + (size_t)head * HDIM;
    const unsigned ldsMsh  = (unsigned)(unsigned long long)(void*)Msh;

    for (int kv0 = 0; kv0 < SEQ; kv0 += KT) {
        // ---- TDM: mask tile (128 x 64 f32) global -> LDS, wave 0 issues ----
        if (wave == 0) {
            const unsigned long long gaddr =
                (unsigned long long)(const void*)(mask + (size_t)(qblk * QBLK) * SEQ + kv0);
            u32x4 g0;
            g0[0] = 1u;                                        // count=1, user D#
            g0[1] = ldsMsh;                                    // lds_addr
            g0[2] = (unsigned)gaddr;                           // global_addr[31:0]
            g0[3] = ((unsigned)(gaddr >> 32) & 0x01FFFFFFu)    // global_addr[56:32]
                    | (2u << 30);                              // type=2 (image)
            i32x8 g1;
            g1[0] = (int)(2u << 16);                           // data_size=4B
            g1[1] = (int)(((unsigned)SEQ) << 16);              // tensor_dim0[15:0]=2048
            g1[2] = (int)(((unsigned)SEQ) << 16);              // tensor_dim1[15:0]=2048
            g1[3] = (int)(((unsigned)KT) << 16);               // tile_dim0=64
            g1[4] = (int)QBLK;                                 // tile_dim1=128, tile_dim2=0
            g1[5] = (int)SEQ;                                  // tensor_dim0_stride=2048
            g1[6] = 0;
            g1[7] = 0;
            const i32x4 gz = {};
#if __clang_major__ >= 23
            const i32x8 gz8 = {};
            __builtin_amdgcn_tensor_load_to_lds(g0, g1, gz, gz, gz8, 0);
#else
            __builtin_amdgcn_tensor_load_to_lds(g0, g1, gz, gz, 0);
#endif
            __builtin_amdgcn_s_wait_tensorcnt(0);
        }

        // ---- cooperative stage: K (f16 row-major), V (f16 transposed) ----
        for (int idx = tid; idx < KT * HDIM; idx += 256) {
            const int r = idx >> 7, d = idx & 127;
            const size_t g = kvBase + (size_t)(kv0 + r) * HID + d;
            Ksh[idx]        = (_Float16)K[g];
            Vsh[d * KT + r] = (_Float16)V[g];
            if (kv0 + KT < SEQ) {                 // prefetch next tile into L2
                __builtin_prefetch(&K[g + (size_t)KT * HID], 0, 0);
                __builtin_prefetch(&V[g + (size_t)KT * HID], 0, 0);
            }
        }
        __syncthreads();

        // ---- S = Q @ K^T : four 16x16 score tiles, 4 K-chunks each ----
        v8f s[NT];
        #pragma unroll
        for (int j = 0; j < NT; ++j) s[j] = {};
        #pragma unroll
        for (int c = 0; c < 4; ++c) {
            #pragma unroll
            for (int j = 0; j < NT; ++j) {
                const v16h b = *(const v16h*)&Ksh[(j * 16 + n16) * HDIM + c * 32 + hhalf * 16];
                s[j] = __builtin_amdgcn_wmma_f32_16x16x32_f16(false, qf[c], false, b,
                                                              (short)0, s[j], false, false);
            }
        }

        // ---- mask add + online softmax (max via lane-group tree, sum via WMMA) ----
        float corr[8];
        #pragma unroll
        for (int r = 0; r < 8; ++r) {
            const int mq = wave * QT + r + 8 * hhalf;
            float e[NT];
            float tmax = -1e30f;
            #pragma unroll
            for (int j = 0; j < NT; ++j) {
                e[j] = s[j][r] + Msh[mq * KT + j * 16 + n16];
                tmax = fmaxf(tmax, e[j]);
            }
            #pragma unroll
            for (int off = 1; off < 16; off <<= 1)
                tmax = fmaxf(tmax, __shfl_xor(tmax, off, 32));

            const float mnew = fmaxf(mrow[r], tmax);
            corr[r] = __expf(mrow[r] - mnew);
            mrow[r] = mnew;

            #pragma unroll
            for (int j = 0; j < NT; ++j) {
                const float p = __expf(e[j] - mnew);
                Psh[wave][(r + 8 * hhalf) * KT + j * 16 + n16] = (_Float16)p;
            }
            #pragma unroll
            for (int t = 0; t < 8; ++t) o[t][r] *= corr[r];
        }

        // wave-private LDS: DS ops are in-order per wave; wait then reload
        asm volatile("s_wait_dscnt 0" ::: "memory");

        // ---- P as two 16x32 A-fragments ----
        v16h pa[2];
        #pragma unroll
        for (int c = 0; c < 2; ++c) {
            const _Float16* prow = &Psh[wave][n16 * KT + c * 32];
            const v8h lo = *(const v8h*)&prow[hhalf * 16];
            const v8h hi = *(const v8h*)&prow[hhalf * 16 + 8];
            v16h a = {};
            #pragma unroll
            for (int i = 0; i < 8; ++i) { a[i] = lo[i]; a[8 + i] = hi[i]; }
            pa[c] = a;
        }

        // ---- row sums of P via WMMA against all-ones B (replicated per lane) ----
        v8f rowsum = {};
        rowsum = __builtin_amdgcn_wmma_f32_16x16x32_f16(false, pa[0], false, vones,
                                                        (short)0, rowsum, false, false);
        rowsum = __builtin_amdgcn_wmma_f32_16x16x32_f16(false, pa[1], false, vones,
                                                        (short)0, rowsum, false, false);
        #pragma unroll
        for (int r = 0; r < 8; ++r)
            lrow[r] = lrow[r] * corr[r] + rowsum[r];

        // ---- O += P @ V : 8 d-tiles x 2 K-chunks ----
        #pragma unroll
        for (int t = 0; t < 8; ++t) {
            #pragma unroll
            for (int c = 0; c < 2; ++c) {
                const v16h bv = *(const v16h*)&Vsh[(t * 16 + n16) * KT + c * 32 + hhalf * 16];
                o[t] = __builtin_amdgcn_wmma_f32_16x16x32_f16(false, pa[c], false, bv,
                                                              (short)0, o[t], false, false);
            }
        }
        __syncthreads();
    }

    // ---- epilogue: normalize by row sum and store ----
    #pragma unroll
    for (int r = 0; r < 8; ++r) {
        const float inv = 1.0f / lrow[r];
        const int   row = q0 + r + 8 * hhalf;
        float* orow = out + ((size_t)(batch * SEQ + row)) * HID + head * HDIM;
        #pragma unroll
        for (int t = 0; t < 8; ++t)
            orow[t * 16 + n16] = o[t][r] * inv;
    }
}

extern "C" void kernel_launch(void* const* d_in, const int* in_sizes, int n_in,
                              void* d_out, int out_size, void* d_ws, size_t ws_size,
                              hipStream_t stream) {
    const float* q    = (const float*)d_in[0];
    const float* k    = (const float*)d_in[1];
    const float* v    = (const float*)d_in[2];
    const float* mask = (const float*)d_in[3];
    float*       out  = (float*)d_out;

    const int blocks = 2 /*batch*/ * HEADS * (SEQ / QBLK);   // 512
    fa_fwd_f16wmma<<<blocks, 256, 0, stream>>>(q, k, v, mask, out);
}
